// ModulatedConv2d_58909771432208
// MI455X (gfx1250) — matrix operand order repaired and checked by reading, not compiled, not run on hardware
//
#include <hip/hip_runtime.h>
#include <hip/hip_bf16.h>
#include <stddef.h>

// ---------------------------------------------------------------- types
typedef __attribute__((ext_vector_type(16))) __bf16 v16bf;
typedef __attribute__((ext_vector_type(8)))  float  v8f;

#define B_    16
#define CIN_  512
#define COUT_ 512
#define HW_   64
#define EPS_  1e-8f

__device__ __forceinline__ unsigned short f2bf(float f) {
    unsigned int u = __float_as_uint(f);
    unsigned int r = (u + 0x7FFFu + ((u >> 16) & 1u)) >> 16;   // RNE
    return (unsigned short)r;
}

// ------------------------------------------------- 1) s = style@mod_w^T + mod_b
__global__ __launch_bounds__(256) void style_kernel(
    const float* __restrict__ style, const float* __restrict__ mod_w,
    const float* __restrict__ mod_b, float* __restrict__ s)
{
    int tid = blockIdx.x * 256 + threadIdx.x;          // 16*512 = 8192
    if (tid >= B_ * CIN_) return;
    int b = tid >> 9, cin = tid & 511;
    float acc = mod_b[cin];
    const float* st = style + (size_t)b * 512;
    const float* mw = mod_w + (size_t)cin * 512;
    #pragma unroll 4
    for (int k = 0; k < 512; ++k) acc += st[k] * mw[k];
    s[tid] = acc;
}

// ------------------------------------------------- 2) demod[b][cout]
__global__ __launch_bounds__(256) void demod_kernel(
    const float* __restrict__ weight, const float* __restrict__ s,
    float* __restrict__ demod)
{
    int tid = blockIdx.x * 256 + threadIdx.x;          // 16*512 = 8192
    if (tid >= B_ * COUT_) return;
    int b = tid >> 9, cout = tid & 511;
    float acc = 0.f;
    for (int cin = 0; cin < CIN_; ++cin) {
        float sv = s[b * CIN_ + cin];
        float sv2 = sv * sv;
        const float* wp = weight + ((size_t)cout * CIN_ + cin) * 9;
        #pragma unroll
        for (int t = 0; t < 9; ++t) { float w = wp[t]; acc += w * w * sv2; }
    }
    demod[tid] = rsqrtf(acc + EPS_);
}

// ------------------------------------------------- 3) bf16 weights, A-fragment swizzled
// layout: wsA[((((b*9+tap)*32 + mt)*16 + kc)*32 + lane)*16 + e]
// lane: M = lane&15, half = lane>>4
// e:    K = half*16 + e
__global__ __launch_bounds__(256) void swizzle_kernel(
    const float* __restrict__ weight, const float* __restrict__ s,
    const float* __restrict__ demod, unsigned short* __restrict__ wsA)
{
    size_t fid = (size_t)blockIdx.x * 256 + threadIdx.x;   // 37,748,736 total
    int e    = (int)(fid & 15);
    int lane = (int)((fid >> 4) & 31);
    int kc   = (int)((fid >> 9) & 15);
    int mt   = (int)((fid >> 13) & 31);
    int bt   = (int)(fid >> 18);                           // b*9 + tap
    int tap  = bt % 9;
    int b    = bt / 9;

    int M = lane & 15, half = lane >> 4;
    int g = e >> 1, p = e & 1;
    int K = half * 16 + g * 2 + p;
    int cout = mt * 16 + M;
    int cin  = kc * 32 + K;
    int ky = tap / 3, kx = tap % 3;

    float v = weight[(((size_t)cout * CIN_ + cin) * 3 + ky) * 3 + kx]
            * s[b * CIN_ + cin] * demod[b * COUT_ + cout];
    wsA[fid] = f2bf(v);
}

// ------------------------------------------------- 4) implicit-GEMM conv via WMMA
// block = 256 COUT x 64 px (one image row), 8 waves, wave = 32(M) x 64(N) = 2x4 wmma tiles
__global__ __launch_bounds__(256) void conv_kernel(
    const float* __restrict__ x, const unsigned short* __restrict__ wsA,
    const float* __restrict__ bias, float* __restrict__ out)
{
    const int tid  = threadIdx.x;
    const int lane = tid & 31;
    const int wave = tid >> 5;               // 0..7 -> M strip

    int bid = blockIdx.x;
    const int mtile = bid & 1;  bid >>= 1;   // COUT half (0..1)
    const int y     = bid & 63; bid >>= 6;   // output row
    const int b     = bid;                   // batch

    // staged input: rows y-1..y+1, px -1..64, 32 cin of current chunk (bf16)
    __shared__ alignas(16) unsigned short xs[3][66][40];   // pad 40 -> 16B-aligned frags

    v8f acc[2][4] = {};

    const int coutBase = mtile * 256 + wave * 32;
    const int nlane    = lane & 15;
    const int half     = lane >> 4;

    for (int kc = 0; kc < 16; ++kc) {
        __syncthreads();
        // ---- stage 3*66*32 = 6336 bf16 elements (coalesced over px)
        for (int f = tid; f < 6336; f += 256) {
            int px = f % 66;
            int rc = f / 66;
            int c  = rc & 31;
            int r  = rc >> 5;
            int yi = y + r - 1;
            int gx = px - 1;
            float v = 0.f;
            if ((unsigned)yi < 64u && (unsigned)gx < 64u)
                v = x[(((size_t)b * CIN_ + (kc * 32 + c)) * HW_ + yi) * HW_ + gx];
            xs[r][px][c] = f2bf(v);
        }
        // prefetch next cin-chunk's center row into cache
        if (kc + 1 < 16 && tid < 64)
            __builtin_prefetch(&x[(((size_t)b * CIN_ + ((kc + 1) * 32)) * HW_ + y) * HW_ + tid], 0, 3);
        __syncthreads();

        for (int tap = 0; tap < 9; ++tap) {
            const int ky = tap / 3, kx = tap % 3;

            // A fragments: pre-swizzled, one 32B load each (unique per wave, no dup)
            size_t aoff = ((((size_t)(b * 9 + tap) * 32 + (coutBase >> 4)) * 16 + kc) * 32 + lane) * 16;
            v16bf a0 = *(const v16bf*)(wsA + aoff);
            v16bf a1 = *(const v16bf*)(wsA + aoff + 16 * 32 * 16);   // next 16-row M tile

            // B fragments from LDS: lane N = px, 16 contiguous cin (K = half*16 + e)
            const unsigned short* bp = &xs[ky][nlane + kx][half * 16];
            v16bf bf0 = *(const v16bf*)(bp);
            v16bf bf1 = *(const v16bf*)(bp + 16 * 40);               // px + 16
            v16bf bf2 = *(const v16bf*)(bp + 32 * 40);               // px + 32
            v16bf bf3 = *(const v16bf*)(bp + 48 * 40);               // px + 48

            acc[0][0] = __builtin_amdgcn_wmma_f32_16x16x32_bf16(false, a0, false, bf0, (short)0, acc[0][0], false, false);
            acc[0][1] = __builtin_amdgcn_wmma_f32_16x16x32_bf16(false, a0, false, bf1, (short)0, acc[0][1], false, false);
            acc[0][2] = __builtin_amdgcn_wmma_f32_16x16x32_bf16(false, a0, false, bf2, (short)0, acc[0][2], false, false);
            acc[0][3] = __builtin_amdgcn_wmma_f32_16x16x32_bf16(false, a0, false, bf3, (short)0, acc[0][3], false, false);
            acc[1][0] = __builtin_amdgcn_wmma_f32_16x16x32_bf16(false, a1, false, bf0, (short)0, acc[1][0], false, false);
            acc[1][1] = __builtin_amdgcn_wmma_f32_16x16x32_bf16(false, a1, false, bf1, (short)0, acc[1][1], false, false);
            acc[1][2] = __builtin_amdgcn_wmma_f32_16x16x32_bf16(false, a1, false, bf2, (short)0, acc[1][2], false, false);
            acc[1][3] = __builtin_amdgcn_wmma_f32_16x16x32_bf16(false, a1, false, bf3, (short)0, acc[1][3], false, false);
        }
    }

    // ---- epilogue: C/D layout (lanes 0-15: M=r, N=lane; lanes 16-31: M=8+r)
    const int mh = half * 8;
    #pragma unroll
    for (int i = 0; i < 2; ++i) {
        #pragma unroll
        for (int j = 0; j < 4; ++j) {
            int xcol = j * 16 + nlane;
            #pragma unroll
            for (int r = 0; r < 8; ++r) {
                int co = coutBase + i * 16 + mh + r;
                out[(((size_t)b * COUT_ + co) * HW_ + y) * HW_ + xcol] = acc[i][j][r] + bias[co];
            }
        }
    }
}

// ---------------------------------------------------------------- launcher
extern "C" void kernel_launch(void* const* d_in, const int* in_sizes, int n_in,
                              void* d_out, int out_size, void* d_ws, size_t ws_size,
                              hipStream_t stream) {
    const float* x      = (const float*)d_in[0];
    const float* style  = (const float*)d_in[1];
    const float* weight = (const float*)d_in[2];
    const float* bias   = (const float*)d_in[3];
    const float* mod_w  = (const float*)d_in[4];
    const float* mod_b  = (const float*)d_in[5];
    float* out = (float*)d_out;

    float*          s     = (float*)d_ws;                              // 16*512 f32
    float*          demod = s + B_ * CIN_;                             // 16*512 f32
    unsigned short* wsA   = (unsigned short*)((char*)d_ws + 65536);    // 75.5 MB bf16

    style_kernel <<<(B_ * CIN_ + 255) / 256, 256, 0, stream>>>(style, mod_w, mod_b, s);
    demod_kernel <<<(B_ * COUT_ + 255) / 256, 256, 0, stream>>>(weight, s, demod);
    swizzle_kernel<<<147456, 256, 0, stream>>>(weight, s, demod, wsA);   // 37.75M elems
    conv_kernel  <<<B_ * HW_ * (COUT_ / 256), 256, 0, stream>>>(x, wsA, bias, out);
}
